// SelfAttention_4363686773113
// MI455X (gfx1250) — matrix operand order repaired and checked by reading, not compiled, not run on hardware
//
#include <hip/hip_runtime.h>
#include <hip/hip_bf16.h>

// Self-attention (out, attn) for B=1, S=4096, D=512, H=8, d=64 on gfx1250.
// GEMMs: v_wmma_f32_16x16x32_f16, LDS-staged via global_load_async_to_lds_b128.
// Softmax fused: scores emits per-tile (max, expsum) stats; AV kernel applies
// exp(s-M)/Z while reading scores and writes normalized attn to d_out.
// d_out = [ out: S*D f32 | attn: H*S*S f32 ].

typedef __attribute__((ext_vector_type(16))) _Float16 v16h;
typedef __attribute__((ext_vector_type(8)))  _Float16 v8h;
typedef __attribute__((ext_vector_type(8)))  float    v8f;

#define S_LEN 4096
#define D_DIM 512
#define HEADS 8
#define HEAD_DIM 64
#define KTILES (S_LEN / 64)
#define NEGV -1e9f

// ---------------------------------------------------------------------------
// WMMA fragments from LDS tiles (CDNA5 ISA layouts)
// A tile [64 rows][32 K] f16: lane l row = l&15 (+16*waveM), g = l>>4,
//   elems 0..7 = K 8g..8g+7, elems 8..15 = K 16+8g..16+8g+7
// B tile stored [64 N][32 K] f16: lane l N = l&15, elems 0..7 = K 8g..8g+7, elems 8..15 = K 16+8g..16+8g+7
// ---------------------------------------------------------------------------
__device__ __forceinline__ v16h frag_a(const _Float16* As, int row, int g) {
    v8h lo = *(const v8h*)&As[row * 32 + 8 * g];
    v8h hi = *(const v8h*)&As[row * 32 + 16 + 8 * g];
    v16h a;
#pragma unroll
    for (int i = 0; i < 8; ++i) { a[i] = lo[i]; a[i + 8] = hi[i]; }
    return a;
}

__device__ __forceinline__ v16h frag_b(const _Float16* Bs, int n, int g) {
    v8h lo = *(const v8h*)&Bs[n * 32 + 8 * g];
    v8h hi = *(const v8h*)&Bs[n * 32 + 16 + 8 * g];
    v16h b;
#pragma unroll
    for (int i = 0; i < 8; ++i) { b[i] = lo[i]; b[i + 8] = hi[i]; }
    return b;
}

__device__ __forceinline__ v8f wmma_f16(v16h a, v16h b, v8f c) {
    return __builtin_amdgcn_wmma_f32_16x16x32_f16(false, a, false, b,
                                                  (short)0, c, false, false);
}

// Async global->LDS copy of 32 bytes (per-lane), tracked by ASYNCcnt.
// LDS flat address truncated to 32 bits == LDS byte offset (aperture layout).
__device__ __forceinline__ void async_stage32B(const _Float16* gsrc, _Float16* ldst) {
    unsigned l = (unsigned)(uintptr_t)ldst;
    asm volatile("global_load_async_to_lds_b128 %0, %1, off\n\t"
                 "global_load_async_to_lds_b128 %0, %1, off offset:16"
                 :: "v"(l), "v"(gsrc) : "memory");
}

__device__ __forceinline__ void async_wait0() {
    asm volatile("s_wait_asynccnt 0" ::: "memory");
}

// ---------------------------------------------------------------------------
// Weight transpose+convert: src f32 [R][C] -> dst f16 [C][R]
// ---------------------------------------------------------------------------
__global__ __launch_bounds__(256) void transpose_f32_to_f16(
    const float* __restrict__ src, _Float16* __restrict__ dst, int R, int C) {
    int idx = blockIdx.x * 256 + threadIdx.x;
    if (idx >= R * C) return;
    int r = idx / C, c = idx % C;
    dst[(size_t)c * R + r] = (_Float16)src[idx];
}

__global__ __launch_bounds__(256) void transpose_f16(
    const _Float16* __restrict__ src, _Float16* __restrict__ dst, int R, int C) {
    int idx = blockIdx.x * 256 + threadIdx.x;
    if (idx >= R * C) return;
    int r = idx / C, c = idx % C;
    dst[(size_t)c * R + r] = src[idx];
}

// ---------------------------------------------------------------------------
// LayerNorm: one block (256 threads) per row of 512; write f16 xn.
// ---------------------------------------------------------------------------
__global__ __launch_bounds__(256) void layernorm_kernel(
    const float* __restrict__ x, const float* __restrict__ g,
    const float* __restrict__ b, _Float16* __restrict__ xn) {
    const int row = blockIdx.x;
    const int t = threadIdx.x;
    const float* xr = x + (size_t)row * D_DIM;
    float v0 = xr[t], v1 = xr[t + 256];

    __shared__ float red[8];
    const int lane = t & 31, wv = t >> 5;

    float s = v0 + v1;
#pragma unroll
    for (int o = 16; o; o >>= 1) s += __shfl_xor(s, o, 32);
    if (lane == 0) red[wv] = s;
    __syncthreads();
    s = red[0];
#pragma unroll
    for (int j = 1; j < 8; ++j) s += red[j];
    float mean = s / (float)D_DIM;

    float d0 = v0 - mean, d1 = v1 - mean;
    float q = d0 * d0 + d1 * d1;
#pragma unroll
    for (int o = 16; o; o >>= 1) q += __shfl_xor(q, o, 32);
    __syncthreads();
    if (lane == 0) red[wv] = q;
    __syncthreads();
    q = red[0];
#pragma unroll
    for (int j = 1; j < 8; ++j) q += red[j];
    float rstd = rsqrtf(q / (float)D_DIM + 1e-5f);

    xn[(size_t)row * D_DIM + t]       = (_Float16)(d0 * rstd * g[t] + b[t]);
    xn[(size_t)row * D_DIM + t + 256] = (_Float16)(d1 * rstd * g[t + 256] + b[t + 256]);
}

// ---------------------------------------------------------------------------
// Generic WMMA GEMM: C = A[M][K] @ Wt[N][K]^T + bias.
// Block 128 (4 waves); tile 64x64; K step 32; async LDS staging.
// ---------------------------------------------------------------------------
__global__ __launch_bounds__(128) void gemm_f16(
    const _Float16* __restrict__ A, const _Float16* __restrict__ Wt,
    const float* __restrict__ bias, _Float16* __restrict__ Cf16,
    float* __restrict__ Cf32, int M, int N, int K) {
    __shared__ _Float16 As[64 * 32];
    __shared__ _Float16 Bs[64 * 32];
    const int tid = threadIdx.x;
    const int lane = tid & 31, wave = tid >> 5;
    const int g = lane >> 4, lr = lane & 15;
    const int m0 = blockIdx.y * 64, n0 = blockIdx.x * 64;
    const int srow = tid >> 1, scol = (tid & 1) * 16;

    v8f acc[4];
#pragma unroll
    for (int nb = 0; nb < 4; ++nb)
#pragma unroll
        for (int i = 0; i < 8; ++i) acc[nb][i] = 0.0f;

    for (int kk = 0; kk < K; kk += 32) {
        if (kk + 32 < K)  // speculative prefetch of next A tile row
            __builtin_prefetch(&A[(size_t)(m0 + srow) * K + kk + 32], 0, 1);
        async_stage32B(&A[(size_t)(m0 + srow) * K + kk + scol], &As[srow * 32 + scol]);
        async_stage32B(&Wt[(size_t)(n0 + srow) * K + kk + scol], &Bs[srow * 32 + scol]);
        async_wait0();
        __syncthreads();
        v16h a = frag_a(As, wave * 16 + lr, g);
#pragma unroll
        for (int nb = 0; nb < 4; ++nb) {
            v16h b = frag_b(Bs, nb * 16 + lr, g);
            acc[nb] = wmma_f16(a, b, acc[nb]);
        }
        __syncthreads();
    }

#pragma unroll
    for (int nb = 0; nb < 4; ++nb) {
        int n = n0 + nb * 16 + lr;
        float bz = bias ? bias[n] : 0.0f;
#pragma unroll
        for (int r = 0; r < 8; ++r) {
            int m = m0 + wave * 16 + r + 8 * g;
            float val = acc[nb][r] + bz;
            if (Cf16) Cf16[(size_t)m * N + n] = (_Float16)val;
            if (Cf32) Cf32[(size_t)m * N + n] = val;
        }
    }
}

// ---------------------------------------------------------------------------
// Scores: q.kT/sqrt(d) + ALiBi + mask -> attn (raw, pre-softmax) and
// per-(row, 64k-tile) partial softmax stats (tile max, tile expsum).
// ---------------------------------------------------------------------------
__global__ __launch_bounds__(128) void scores_kernel(
    const _Float16* __restrict__ Q, const _Float16* __restrict__ Km,
    const float* __restrict__ coords, const unsigned char* __restrict__ am,
    const unsigned char* __restrict__ alm, float* __restrict__ attn,
    float2* __restrict__ stats) {
    const int h = blockIdx.z;
    const int q0 = blockIdx.y * 64, k0 = blockIdx.x * 64;
    __shared__ _Float16 As[64 * 32];
    __shared__ _Float16 Bs[64 * 32];
    __shared__ float cQ[128], cK[128];
    const int tid = threadIdx.x;
    const int lane = tid & 31, wave = tid >> 5;
    const int g = lane >> 4, lr = lane & 15;
    const int srow = tid >> 1, scol = (tid & 1) * 16;

    if (tid < 64) {
        cQ[tid * 2]     = coords[(size_t)(q0 + tid) * 2];
        cQ[tid * 2 + 1] = coords[(size_t)(q0 + tid) * 2 + 1];
    } else {
        int t2 = tid - 64;
        cK[t2 * 2]     = coords[(size_t)(k0 + t2) * 2];
        cK[t2 * 2 + 1] = coords[(size_t)(k0 + t2) * 2 + 1];
    }

    v8f acc[4];
#pragma unroll
    for (int nb = 0; nb < 4; ++nb)
#pragma unroll
        for (int i = 0; i < 8; ++i) acc[nb][i] = 0.0f;

#pragma unroll
    for (int kk = 0; kk < HEAD_DIM; kk += 32) {
        async_stage32B(&Q[(size_t)(q0 + srow) * D_DIM + h * HEAD_DIM + kk + scol],
                       &As[srow * 32 + scol]);
        async_stage32B(&Km[(size_t)(k0 + srow) * D_DIM + h * HEAD_DIM + kk + scol],
                       &Bs[srow * 32 + scol]);
        async_wait0();
        __syncthreads();
        v16h a = frag_a(As, wave * 16 + lr, g);
#pragma unroll
        for (int nb = 0; nb < 4; ++nb) {
            v16h b = frag_b(Bs, nb * 16 + lr, g);
            acc[nb] = wmma_f16(a, b, acc[nb]);
        }
        __syncthreads();
    }

    const float scale = 0.125f;                  // 1/sqrt(64)
    const float slope = exp2f(-(float)(h + 1));  // 2^(-8*(h+1)/H), H=8
    float* out = attn + (size_t)h * S_LEN * S_LEN;
#pragma unroll
    for (int r = 0; r < 8; ++r) {
        const int ql = wave * 16 + r + 8 * g;
        const int qi = q0 + ql;
        float sc4[4];
        float lmax = -3.4e38f;
#pragma unroll
        for (int nb = 0; nb < 4; ++nb) {
            int kl = nb * 16 + lr;
            int ki = k0 + kl;
            float sc = acc[nb][r] * scale;
            float dx = cQ[ql * 2] - cK[kl * 2];
            float dy = cQ[ql * 2 + 1] - cK[kl * 2 + 1];
            float dist = sqrtf(dx * dx + dy * dy + 1e-12f);
            size_t idx = (size_t)qi * S_LEN + ki;
            sc += -slope * dist * (alm[idx] ? 1.0f : 0.0f);
            if (am[idx]) sc = NEGV;
            out[idx] = sc;
            sc4[nb] = sc;
            lmax = fmaxf(lmax, sc);
        }
        // reduce across the 16 lanes (lr) holding this row's 64 keys
#pragma unroll
        for (int o = 8; o; o >>= 1) lmax = fmaxf(lmax, __shfl_xor(lmax, o, 32));
        float lsum = 0.0f;
#pragma unroll
        for (int nb = 0; nb < 4; ++nb) lsum += __expf(sc4[nb] - lmax);
#pragma unroll
        for (int o = 8; o; o >>= 1) lsum += __shfl_xor(lsum, o, 32);
        if (lr == 0)
            stats[(size_t)(h * S_LEN + qi) * KTILES + blockIdx.x] =
                make_float2(lmax, lsum);
    }
}

// ---------------------------------------------------------------------------
// Combine per-tile stats -> per-row (M, 1/Z). One 64-thread block per row.
// ---------------------------------------------------------------------------
__global__ __launch_bounds__(64) void softmax_stats_kernel(
    const float2* __restrict__ stats, float2* __restrict__ rowstats) {
    const int row = blockIdx.x;           // h*S + q
    const int t = threadIdx.x;            // 0..63 (2 waves)
    const int lane = t & 31, wv = t >> 5;
    __shared__ float sm[2], ss[2];

    float2 p = stats[(size_t)row * KTILES + t];
    float m = p.x;
#pragma unroll
    for (int o = 16; o; o >>= 1) m = fmaxf(m, __shfl_xor(m, o, 32));
    if (lane == 0) sm[wv] = m;
    __syncthreads();
    float M = fmaxf(sm[0], sm[1]);

    float z = p.y * __expf(p.x - M);
#pragma unroll
    for (int o = 16; o; o >>= 1) z += __shfl_xor(z, o, 32);
    if (lane == 0) ss[wv] = z;
    __syncthreads();
    if (t == 0) rowstats[row] = make_float2(M, 1.0f / (ss[0] + ss[1]));
}

// ---------------------------------------------------------------------------
// attn @ V with fused softmax normalization: reads raw scores, computes
// p = exp(s - M) / Z, writes p back to attn (d_out), uses f16 p for WMMA.
// ---------------------------------------------------------------------------
__global__ __launch_bounds__(128) void av_kernel(
    float* __restrict__ attn, const float2* __restrict__ rowstats,
    const _Float16* __restrict__ Vt, _Float16* __restrict__ OutH) {
    const int h = blockIdx.y;
    const int q0 = blockIdx.x * 64;
    __shared__ _Float16 As[64 * 32];
    __shared__ _Float16 Bs[64 * 32];
    const int tid = threadIdx.x;
    const int lane = tid & 31, wave = tid >> 5;
    const int g = lane >> 4, lr = lane & 15;
    const int srow = tid >> 1, scol = (tid & 1) * 16;
    float* Ah = attn + (size_t)h * S_LEN * S_LEN;

    // per-staging-thread row statistics (fixed row per thread)
    const float2 rs = rowstats[(size_t)h * S_LEN + q0 + srow];
    float* arow = &Ah[(size_t)(q0 + srow) * S_LEN];

    v8f acc[4];
#pragma unroll
    for (int nb = 0; nb < 4; ++nb)
#pragma unroll
        for (int i = 0; i < 8; ++i) acc[nb][i] = 0.0f;

    for (int kk = 0; kk < S_LEN; kk += 32) {
        {
            float4* src = (float4*)&arow[kk + scol];
            float fv[16];
            *(float4*)&fv[0]  = src[0];
            *(float4*)&fv[4]  = src[1];
            *(float4*)&fv[8]  = src[2];
            *(float4*)&fv[12] = src[3];
#pragma unroll
            for (int i = 0; i < 16; ++i) fv[i] = __expf(fv[i] - rs.x) * rs.y;
            src[0] = *(float4*)&fv[0];   // normalized attn back to d_out
            src[1] = *(float4*)&fv[4];
            src[2] = *(float4*)&fv[8];
            src[3] = *(float4*)&fv[12];
            v8h h0, h1;
#pragma unroll
            for (int i = 0; i < 8; ++i) {
                h0[i] = (_Float16)fv[i];
                h1[i] = (_Float16)fv[8 + i];
            }
            *(v8h*)&As[srow * 32 + scol] = h0;
            *(v8h*)&As[srow * 32 + scol + 8] = h1;
        }
        async_stage32B(&Vt[(size_t)(h * HEAD_DIM + srow) * S_LEN + kk + scol],
                       &Bs[srow * 32 + scol]);
        async_wait0();
        __syncthreads();
        v16h a = frag_a(As, wave * 16 + lr, g);
#pragma unroll
        for (int nb = 0; nb < 4; ++nb) {
            v16h b = frag_b(Bs, nb * 16 + lr, g);
            acc[nb] = wmma_f16(a, b, acc[nb]);
        }
        __syncthreads();
    }

#pragma unroll
    for (int nb = 0; nb < 4; ++nb) {
        int dd = nb * 16 + lr;
#pragma unroll
        for (int r = 0; r < 8; ++r) {
            int qi = q0 + wave * 16 + r + 8 * g;
            OutH[(size_t)qi * D_DIM + h * HEAD_DIM + dd] = (_Float16)acc[nb][r];
        }
    }
}

// ---------------------------------------------------------------------------
extern "C" void kernel_launch(void* const* d_in, const int* in_sizes, int n_in,
                              void* d_out, int out_size, void* d_ws, size_t ws_size,
                              hipStream_t stream) {
    const float* x          = (const float*)d_in[0];
    const float* coords     = (const float*)d_in[1];
    const unsigned char* am = (const unsigned char*)d_in[2];  // numpy bool = 1B
    const unsigned char* al = (const unsigned char*)d_in[3];
    const float* ln_g = (const float*)d_in[4];
    const float* ln_b = (const float*)d_in[5];
    const float* Wq = (const float*)d_in[6];
    const float* bq = (const float*)d_in[7];
    const float* Wk = (const float*)d_in[8];
    const float* bk = (const float*)d_in[9];
    const float* Wv = (const float*)d_in[10];
    const float* bv = (const float*)d_in[11];
    const float* Wo = (const float*)d_in[12];
    const float* bo = (const float*)d_in[13];

    const int S = S_LEN, D = D_DIM, H = HEADS;
    char* ws = (char*)d_ws;
    size_t off = 0;
    auto alloc = [&](size_t bytes) {
        char* p = ws + off;
        off = (off + bytes + 255) & ~(size_t)255;
        return p;
    };
    _Float16* xn16 = (_Float16*)alloc((size_t)S * D * 2);
    _Float16* q16  = (_Float16*)alloc((size_t)S * D * 2);
    _Float16* k16  = (_Float16*)alloc((size_t)S * D * 2);
    _Float16* v16  = (_Float16*)alloc((size_t)S * D * 2);
    _Float16* vt16 = (_Float16*)alloc((size_t)S * D * 2);
    _Float16* oh16 = (_Float16*)alloc((size_t)S * D * 2);
    _Float16* wqT  = (_Float16*)alloc((size_t)D * D * 2);
    _Float16* wkT  = (_Float16*)alloc((size_t)D * D * 2);
    _Float16* wvT  = (_Float16*)alloc((size_t)D * D * 2);
    _Float16* woT  = (_Float16*)alloc((size_t)D * D * 2);
    float2* stats    = (float2*)alloc((size_t)H * S * KTILES * sizeof(float2));
    float2* rowstats = (float2*)alloc((size_t)H * S * sizeof(float2));

    float* out_f32  = (float*)d_out;               // [S][D]
    float* attn_f32 = out_f32 + (size_t)S * D;     // [H][S][S]

    // 1. weights -> f16 transposed [N][K]
    int wblocks = (D * D + 255) / 256;
    transpose_f32_to_f16<<<wblocks, 256, 0, stream>>>(Wq, wqT, D, D);
    transpose_f32_to_f16<<<wblocks, 256, 0, stream>>>(Wk, wkT, D, D);
    transpose_f32_to_f16<<<wblocks, 256, 0, stream>>>(Wv, wvT, D, D);
    transpose_f32_to_f16<<<wblocks, 256, 0, stream>>>(Wo, woT, D, D);

    // 2. LayerNorm -> xn f16
    layernorm_kernel<<<S, 256, 0, stream>>>(x, ln_g, ln_b, xn16);

    // 3. Q/K/V projections (WMMA)
    dim3 ggrid(D / 64, S / 64);
    gemm_f16<<<ggrid, 128, 0, stream>>>(xn16, wqT, bq, q16, nullptr, S, D, D);
    gemm_f16<<<ggrid, 128, 0, stream>>>(xn16, wkT, bk, k16, nullptr, S, D, D);
    gemm_f16<<<ggrid, 128, 0, stream>>>(xn16, wvT, bv, v16, nullptr, S, D, D);

    // 4. scores + ALiBi + mask -> raw attn + partial softmax stats
    dim3 sgrid(S / 64, S / 64, H);
    scores_kernel<<<sgrid, 128, 0, stream>>>(q16, k16, coords, am, al,
                                             attn_f32, stats);

    // 5. combine stats -> per-row (M, 1/Z)
    softmax_stats_kernel<<<H * S, 64, 0, stream>>>(stats, rowstats);

    // 6. V transpose -> [D][S]
    transpose_f16<<<(S * D + 255) / 256, 256, 0, stream>>>(v16, vt16, S, D);

    // 7. attn @ V with fused softmax normalization (WMMA)
    dim3 agrid(S / 64, H);
    av_kernel<<<agrid, 128, 0, stream>>>(attn_f32, rowstats, vt16, oh16);

    // 8. output projection (WMMA) -> f32 out
    gemm_f16<<<ggrid, 128, 0, stream>>>(oh16, woT, bo, nullptr, out_f32, S, D, D);
}
